// PartContextMamba_25185688223947
// MI455X (gfx1250) — matrix operand order repaired and checked by reading, not compiled, not run on hardware
//
#include <hip/hip_runtime.h>
#include <hip/hip_bf16.h>
#include <math.h>

// ---------------------------------------------------------------------------
// Bidirectional Mamba block for MI455X (gfx1250), wave32 + WMMA bf16 +
// async global->LDS staging (ASYNCcnt) with double buffering.
// Shapes: B=1024, L=6, D_MODEL=768, D_INNER=1536, D_STATE=16, DT_RANK=48.
// M = B*L = 6144 rows for all GEMMs.
// ---------------------------------------------------------------------------

#define BB   1024
#define LL   6
#define DM   768
#define DI   1536
#define N2   3072      // 2*DI
#define NST  16
#define RK   48
#define KD   64        // DT_RANK padded to multiple of 32
#define XD   80        // DT_RANK + 2*D_STATE
#define MR   6144      // BB*LL

typedef __bf16 v16bf __attribute__((ext_vector_type(16)));
typedef float  v8f   __attribute__((ext_vector_type(8)));

__device__ __forceinline__ unsigned short f2bf(float f) {
    unsigned u = __float_as_uint(f);
    unsigned r = u + 0x7FFFu + ((u >> 16) & 1u);   // round-to-nearest-even
    return (unsigned short)(r >> 16);
}

// issue one per-lane async 16B global->LDS copy (ASYNCcnt-tracked, no VGPR data)
__device__ __forceinline__ void async_b128(unsigned lds_off, const void* gaddr) {
    asm volatile("global_load_async_to_lds_b128 %0, %1, off"
                 :: "v"(lds_off), "v"(gaddr) : "memory");
}

// ---------------------------------------------------------------------------
// fp32 -> bf16 casts
// ---------------------------------------------------------------------------
__global__ __launch_bounds__(256)
void cast_bf16_kernel(const float* __restrict__ src, unsigned short* __restrict__ dst, int n) {
    int i = blockIdx.x * 256 + threadIdx.x;
    if (i < n) dst[i] = f2bf(src[i]);
}

// gather kin columns of a strided fp32 matrix, zero-pad to kout, cast to bf16
__global__ __launch_bounds__(256)
void pad_cast_kernel(const float* __restrict__ src, unsigned short* __restrict__ dst,
                     int rows, int src_stride, int kin, int kout) {
    int i = blockIdx.x * 256 + threadIdx.x;
    if (i >= rows * kout) return;
    int r = i / kout, c = i % kout;
    float v = (c < kin) ? src[(size_t)r * src_stride + c] : 0.0f;
    dst[i] = f2bf(v);
}

// ---------------------------------------------------------------------------
// WMMA bf16 GEMM:  C[M,N] (f32) = A[M,K](bf16 rowmajor) * W[N,K](bf16 rowmajor)^T
// act==1: C = softplus(C + bias[col])   (fused dt epilogue)
// Block: 256 thr (8 waves). Tile 128x128x32, double-buffered async LDS staging.
// Wave grid 4(M) x 2(N); each wave computes 32x64 = 2x4 fragments of
// v_wmma_f32_16x16x32_bf16.
// Requires: K % 32 == 0, M % 128 == 0. N may be ragged: output is
// bounds-checked; B-tile rows beyond N load garbage that only feeds masked
// output columns (per-column dataflow of WMMA), so no zero-fill is needed.
// ---------------------------------------------------------------------------
#define LDSW 40   // halves per LDS row (32 data + 8 pad) -> 80B rows, 16B aligned

__global__ __launch_bounds__(256)
void gemm_bf16_kernel(const unsigned short* __restrict__ A,
                      const unsigned short* __restrict__ W,
                      float* __restrict__ C,
                      const float* __restrict__ bias,
                      int M, int N, int K, int act) {
    __shared__ unsigned short lA[2][128 * LDSW];
    __shared__ unsigned short lB[2][128 * LDSW];

    const int tid   = threadIdx.x;
    const int lane  = tid & 31;
    const int wave  = tid >> 5;
    const int waveM = wave & 3;    // 0..3 -> 32-row slabs
    const int waveN = wave >> 2;   // 0..1 -> 64-col slabs
    const int M0    = blockIdx.y * 128;
    const int N0    = blockIdx.x * 128;
    const int h     = lane >> 4;   // lane half-group
    const int l15   = lane & 15;

    // staging assignment: each thread owns 2 chunks (8 halves) per tile
    const int r0 = tid >> 2;            // rows 0..63
    const int r1 = 64 + (tid >> 2);     // rows 64..127
    const int cc = (tid & 3) * 8;       // half offset within row

    v8f zero;
    #pragma unroll
    for (int e = 0; e < 8; e++) zero[e] = 0.0f;
    v8f acc[2][4];
    #pragma unroll
    for (int i = 0; i < 2; i++)
        #pragma unroll
        for (int j = 0; j < 4; j++) acc[i][j] = zero;

    // issue the 4 async b128 copies (A r0/r1, W r0/r1) for k-step k0 into buf
    auto issue_stage = [&](int k0, int buf) {
        async_b128((unsigned)(size_t)(void*)&lA[buf][r0 * LDSW + cc],
                   A + (size_t)(M0 + r0) * K + k0 + cc);
        async_b128((unsigned)(size_t)(void*)&lA[buf][r1 * LDSW + cc],
                   A + (size_t)(M0 + r1) * K + k0 + cc);
        async_b128((unsigned)(size_t)(void*)&lB[buf][r0 * LDSW + cc],
                   W + (size_t)(N0 + r0) * K + k0 + cc);
        async_b128((unsigned)(size_t)(void*)&lB[buf][r1 * LDSW + cc],
                   W + (size_t)(N0 + r1) * K + k0 + cc);
    };

    const int S = K / 32;
    issue_stage(0, 0);

    for (int s = 0; s < S; s++) {
        const int buf = s & 1;
        if (s + 1 < S) {
            issue_stage((s + 1) * 32, buf ^ 1);          // prefetch next stage
            asm volatile("s_wait_asynccnt 0x4" ::: "memory");  // stage s landed
        } else {
            asm volatile("s_wait_asynccnt 0x0" ::: "memory");
        }
        __syncthreads();

        // ---- B fragments: lane l15 = column, 16 contiguous K halves @ K=16h ----
        v16bf bfrag[4];
        #pragma unroll
        for (int nt = 0; nt < 4; nt++) {
            int n = waveN * 64 + nt * 16 + l15;
            const uint4* p = reinterpret_cast<const uint4*>(&lB[buf][n * LDSW + h * 16]);
            union { uint4 q[2]; v16bf v; } u;
            u.q[0] = p[0];
            u.q[1] = p[1];
            bfrag[nt] = u.v;
        }
        // ---- A fragments + WMMA ----
        #pragma unroll
        for (int mt = 0; mt < 2; mt++) {
            int m = waveM * 32 + mt * 16 + l15;
            const unsigned short* rowp = &lA[buf][m * LDSW];
            union { unsigned int d[8]; v16bf v; } ua;
            #pragma unroll
            for (int p = 0; p < 4; p++)
                ua.d[p] = *reinterpret_cast<const unsigned int*>(rowp + h * 16 + 2 * p);
            #pragma unroll
            for (int p = 0; p < 4; p++)
                ua.d[4 + p] = *reinterpret_cast<const unsigned int*>(rowp + h * 16 + 8 + 2 * p);
            v16bf afrag = ua.v;
            #pragma unroll
            for (int nt = 0; nt < 4; nt++)
                acc[mt][nt] = __builtin_amdgcn_wmma_f32_16x16x32_bf16(
                    false, afrag, false, bfrag[nt], (short)0, acc[mt][nt], false, false);
        }
        __syncthreads();   // all waves done reading buf before it is re-filled
    }

    // ---- epilogue / store: frag elem r -> row = base + r + 8h, col = lane&15 ----
    #pragma unroll
    for (int mt = 0; mt < 2; mt++) {
        #pragma unroll
        for (int nt = 0; nt < 4; nt++) {
            int col = N0 + waveN * 64 + nt * 16 + l15;
            if (col >= N) continue;
            int rowBase = M0 + waveM * 32 + mt * 16 + 8 * h;
            float bv = bias ? bias[col] : 0.0f;
            #pragma unroll
            for (int r = 0; r < 8; r++) {
                float v = acc[mt][nt][r];
                if (act == 1) {
                    v += bv;
                    v = (v > 20.0f) ? v : logf(1.0f + expf(v));  // softplus
                }
                C[(size_t)(rowBase + r) * N + col] = v;
            }
        }
    }
}

// ---------------------------------------------------------------------------
// causal depthwise conv (kernel 4) over L=6 + SiLU. rev selects time order for
// the backward direction; data stays in original L-order in memory.
// xz: (MR, N2); channel d is column d of the first DI columns.
// ---------------------------------------------------------------------------
__global__ __launch_bounds__(256)
void conv_silu_kernel(const float* __restrict__ xz,
                      const float* __restrict__ cw, const float* __restrict__ cb,
                      float* __restrict__ xc, unsigned short* __restrict__ xcb, int rev) {
    int idx = blockIdx.x * 256 + threadIdx.x;
    if (idx >= BB * DI) return;
    int b = idx / DI, d = idx % DI;
    float w[4];
    #pragma unroll
    for (int k = 0; k < 4; k++) w[k] = cw[k * DI + d];
    float bias = cb[d];
    float s[LL];
    #pragma unroll
    for (int t = 0; t < LL; t++) {
        int lt = rev ? (LL - 1 - t) : t;
        s[t] = xz[((size_t)b * LL + lt) * N2 + d];
    }
    #pragma unroll
    for (int t = 0; t < LL; t++) {
        float a = bias;
        #pragma unroll
        for (int k = 0; k < 4; k++) {
            int j = t - 3 + k;
            if (j >= 0) a += w[k] * s[j];
        }
        float v = a / (1.0f + expf(-a));   // silu
        int lt = rev ? (LL - 1 - t) : t;
        size_t o = ((size_t)b * LL + lt) * DI + d;
        xc[o]  = v;
        xcb[o] = f2bf(v);
    }
}

// ---------------------------------------------------------------------------
// selective scan: one thread per (b,d); h[16] in registers, 6 sequential steps.
// Fuses +u*D, z-gating (silu) and bf16 cast of the out_proj input.
// ---------------------------------------------------------------------------
__global__ __launch_bounds__(256)
void scan_kernel(const float* __restrict__ dt, const float* __restrict__ xdbl,
                 const float* __restrict__ xc, const float* __restrict__ xz,
                 const float* __restrict__ Alog, const float* __restrict__ Dp,
                 unsigned short* __restrict__ yg, int rev) {
    int idx = blockIdx.x * 256 + threadIdx.x;
    if (idx >= BB * DI) return;
    int b = idx / DI, d = idx % DI;
    float A[NST], h[NST];
    #pragma unroll
    for (int n = 0; n < NST; n++) {
        A[n] = -expf(Alog[(size_t)d * NST + n]);
        h[n] = 0.0f;
    }
    float Dv = Dp[d];
    for (int t = 0; t < LL; t++) {
        int lt = rev ? (LL - 1 - t) : t;
        size_t row = (size_t)b * LL + lt;
        float dtv = dt[row * DI + d];
        float u   = xc[row * DI + d];
        float du  = dtv * u;
        float y = 0.0f;
        #pragma unroll
        for (int n = 0; n < NST; n++) {
            float dA = expf(dtv * A[n]);
            h[n] = dA * h[n] + du * xdbl[row * XD + RK + n];
            y += h[n] * xdbl[row * XD + RK + NST + n];
        }
        y += u * Dv;
        float z = xz[row * N2 + DI + d];
        y *= z / (1.0f + expf(-z));        // * silu(z)
        yg[row * DI + d] = f2bf(y);
    }
}

// ---------------------------------------------------------------------------
// residual + LayerNorm over D_MODEL=768. One block per row, 256 thr x 3 elems.
// ---------------------------------------------------------------------------
__global__ __launch_bounds__(256)
void resid_ln_kernel(const float* __restrict__ x, const float* __restrict__ yf,
                     const float* __restrict__ yb, const float* __restrict__ g,
                     const float* __restrict__ bt, float* __restrict__ out) {
    __shared__ float red[256];
    int row = blockIdx.x;
    int tid = threadIdx.x;
    const float* xr = x  + (size_t)row * DM;
    const float* fr = yf + (size_t)row * DM;
    const float* br = yb + (size_t)row * DM;
    float r[3];
    float s = 0.0f;
    #pragma unroll
    for (int i = 0; i < 3; i++) {
        int c = tid + i * 256;
        r[i] = xr[c] + fr[c] + br[c];
        s += r[i];
    }
    red[tid] = s; __syncthreads();
    for (int off = 128; off > 0; off >>= 1) {
        if (tid < off) red[tid] += red[tid + off];
        __syncthreads();
    }
    float mu = red[0] * (1.0f / (float)DM);
    __syncthreads();
    float sq = 0.0f;
    #pragma unroll
    for (int i = 0; i < 3; i++) { float dd = r[i] - mu; sq += dd * dd; }
    red[tid] = sq; __syncthreads();
    for (int off = 128; off > 0; off >>= 1) {
        if (tid < off) red[tid] += red[tid + off];
        __syncthreads();
    }
    float inv = rsqrtf(red[0] * (1.0f / (float)DM) + 1e-5f);
    #pragma unroll
    for (int i = 0; i < 3; i++) {
        int c = tid + i * 256;
        out[(size_t)row * DM + c] = (r[i] - mu) * inv * g[c] + bt[c];
    }
}

// ---------------------------------------------------------------------------
// host launch
// ---------------------------------------------------------------------------
static inline size_t alignup(size_t x) { return (x + 255) & ~(size_t)255; }

extern "C" void kernel_launch(void* const* d_in, const int* in_sizes, int n_in,
                              void* d_out, int out_size, void* d_ws, size_t ws_size,
                              hipStream_t stream) {
    (void)in_sizes; (void)n_in; (void)out_size; (void)ws_size;

    const float* x    = (const float*)d_in[0];
    // per-direction param pointers: f_* at 1..9, b_* at 10..18
    const float* w_in[2]  = { (const float*)d_in[1],  (const float*)d_in[10] };
    const float* cw[2]    = { (const float*)d_in[2],  (const float*)d_in[11] };
    const float* cb[2]    = { (const float*)d_in[3],  (const float*)d_in[12] };
    const float* xp[2]    = { (const float*)d_in[4],  (const float*)d_in[13] };
    const float* dtw[2]   = { (const float*)d_in[5],  (const float*)d_in[14] };
    const float* dtb[2]   = { (const float*)d_in[6],  (const float*)d_in[15] };
    const float* Alog[2]  = { (const float*)d_in[7],  (const float*)d_in[16] };
    const float* Dp[2]    = { (const float*)d_in[8],  (const float*)d_in[17] };
    const float* w_out[2] = { (const float*)d_in[9],  (const float*)d_in[18] };
    const float* ln_g = (const float*)d_in[19];
    const float* ln_b = (const float*)d_in[20];
    float* out = (float*)d_out;

    // ---- workspace layout ----
    char* base = (char*)d_ws;
    size_t cur = 0;
    auto take = [&](size_t bytes) { void* p = base + cur; cur += alignup(bytes); return p; };

    unsigned short* x_bf = (unsigned short*)take((size_t)MR * DM * 2);
    unsigned short *winb[2], *wxpb[2], *wdtb[2], *woutb[2];
    float *xz[2], *xc[2], *xdbl[2], *dtf[2];
    unsigned short *xcb[2], *dtlo[2];
    for (int d = 0; d < 2; d++) {
        winb[d]  = (unsigned short*)take((size_t)N2 * DM * 2);
        wxpb[d]  = (unsigned short*)take((size_t)XD * DI * 2);
        wdtb[d]  = (unsigned short*)take((size_t)DI * KD * 2);
        woutb[d] = (unsigned short*)take((size_t)DM * DI * 2);
    }
    size_t xz0_off = cur;
    for (int d = 0; d < 2; d++) xz[d]   = (float*)take((size_t)MR * N2 * 4);
    for (int d = 0; d < 2; d++) xc[d]   = (float*)take((size_t)MR * DI * 4);
    for (int d = 0; d < 2; d++) xcb[d]  = (unsigned short*)take((size_t)MR * DI * 2);
    for (int d = 0; d < 2; d++) xdbl[d] = (float*)take((size_t)MR * XD * 4);
    for (int d = 0; d < 2; d++) dtlo[d] = (unsigned short*)take((size_t)MR * KD * 2);
    for (int d = 0; d < 2; d++) dtf[d]  = (float*)take((size_t)MR * DI * 4);
    // aliases: gated scan output reuses xcb (dead after x_proj GEMM);
    // out_proj results reuse xz[0] region (dead after the scans).
    unsigned short* yg[2] = { xcb[0], xcb[1] };
    float* yout[2];
    yout[0] = (float*)(base + xz0_off);
    yout[1] = yout[0] + (size_t)MR * DM;

    auto nblk = [](int n) { return (n + 255) / 256; };
    auto gemm = [&](const unsigned short* A, const unsigned short* W, float* C,
                    const float* bias, int M, int N, int K, int act) {
        dim3 grid((N + 127) / 128, M / 128);
        gemm_bf16_kernel<<<grid, 256, 0, stream>>>(A, W, C, bias, M, N, K, act);
    };

    // ---- stage 0: casts ----
    cast_bf16_kernel<<<nblk(MR * DM), 256, 0, stream>>>(x, x_bf, MR * DM);
    for (int d = 0; d < 2; d++) {
        cast_bf16_kernel<<<nblk(N2 * DM), 256, 0, stream>>>(w_in[d], winb[d], N2 * DM);
        cast_bf16_kernel<<<nblk(XD * DI), 256, 0, stream>>>(xp[d], wxpb[d], XD * DI);
        cast_bf16_kernel<<<nblk(DM * DI), 256, 0, stream>>>(w_out[d], woutb[d], DM * DI);
        pad_cast_kernel<<<nblk(DI * KD), 256, 0, stream>>>(dtw[d], wdtb[d], DI, RK, RK, KD);
    }

    // ---- per-direction pipeline (d=0 fwd, d=1 bwd/reversed-time) ----
    for (int d = 0; d < 2; d++) {
        gemm(x_bf, winb[d], xz[d], nullptr, MR, N2, DM, 0);                       // in_proj
        conv_silu_kernel<<<nblk(BB * DI), 256, 0, stream>>>(
            xz[d], cw[d], cb[d], xc[d], xcb[d], d);                               // conv+silu
        gemm(xcb[d], wxpb[d], xdbl[d], nullptr, MR, XD, DI, 0);                   // x_proj
        pad_cast_kernel<<<nblk(MR * KD), 256, 0, stream>>>(
            xdbl[d], dtlo[d], MR, XD, RK, KD);                                    // dt_lo pad
        gemm(dtlo[d], wdtb[d], dtf[d], dtb[d], MR, DI, KD, 1);                    // dt + softplus
        scan_kernel<<<nblk(BB * DI), 256, 0, stream>>>(
            dtf[d], xdbl[d], xc[d], xz[d], Alog[d], Dp[d], yg[d], d);             // scan + gate
    }

    // ---- out_proj (both directions) + residual/LayerNorm ----
    gemm(yg[0], woutb[0], yout[0], nullptr, MR, DM, DI, 0);
    gemm(yg[1], woutb[1], yout[1], nullptr, MR, DM, DI, 0);
    resid_ln_kernel<<<MR, 256, 0, stream>>>(x, yout[0], yout[1], ln_g, ln_b, out);
}
